// Attention_37314675868009
// MI455X (gfx1250) — compile-verified, matrix operand order checked by reading, not run on hardware
//
#include <hip/hip_runtime.h>

// ---------------------------------------------------------------------------
// Attention (B=2, N=2048, DIM=1024, H=16, DH=64) for gfx1250 (CDNA5, wave32).
// All matmuls via v_wmma_f32_16x16x32_bf16. Workspace layout (bf16 elements):
//   xn   [B,N,DIM]        4M elts
//   q    [B,H,N,DH]       4M (pre-scaled by DH^-0.5)
//   k    [B,H,N,DH]       4M
//   vT   [B,H,DH,N]       4M (transposed for contiguous K-loads in P*V)
//   ao   [B,N,INNER]      4M
//   WqB 1M, WkvB 2M, WoB 1M
// Total = 24M bf16 = 48 MiB of d_ws (L2-resident on MI455X's 192MB L2).
// The 256MB rel_pos_bias is streamed with non-temporal loads (dominant BW).
// ---------------------------------------------------------------------------

#define DIMC   1024
#define NSEQ   2048
#define BATCH  2
#define HEADS  16
#define DHEAD  64
#define INNER  1024

typedef __attribute__((ext_vector_type(16))) __bf16 bf16x16;
typedef __attribute__((ext_vector_type(8)))  float  f32x8;
typedef unsigned int u32x4 __attribute__((ext_vector_type(4)));

union FragU { u32x4 q[2]; bf16x16 v; };

__device__ __forceinline__ unsigned short f2bf_bits(float f) {
  unsigned u = __float_as_uint(f);
  unsigned r = u + 0x7FFFu + ((u >> 16) & 1u);   // round-to-nearest-even
  return (unsigned short)(r >> 16);
}

// CDNA5 16-bit A/B fragment (16x32): lane holds row/col (lane&15);
// lanes 0-15 carry K = {base..base+7, base+16..base+23} with base=0,
// lanes 16-31 the same with base=8.  => two 16B chunks at +base, +base+16.
__device__ __forceinline__ bf16x16 load_frag(const unsigned short* rowk, int lane) {
  int base = (lane >> 4) << 3;                   // 0 or 8
  FragU f;
  f.q[0] = *(const u32x4*)(rowk + base);
  f.q[1] = *(const u32x4*)(rowk + base + 16);
  return f.v;
}

__device__ __forceinline__ f32x8 wmma_bf16(bf16x16 a, bf16x16 b, f32x8 c) {
  return __builtin_amdgcn_wmma_f32_16x16x32_bf16(
      /*neg_a=*/false, a, /*neg_b=*/false, b,
      /*c_mod=*/(short)0, c, /*reuse_a=*/false, /*reuse_b=*/false);
}

// ---------------------------------------------------------------------------
// 1) stats over sequence axis (dim=1) + normalize -> bf16 xn
//    one thread per (b, d) column: coalesced across threads at each n.
// ---------------------------------------------------------------------------
__global__ void stats_norm_kernel(const float* __restrict__ x,
                                  const float* __restrict__ g,
                                  unsigned short* __restrict__ xn) {
  int d = blockIdx.x * blockDim.x + threadIdx.x;   // 0..DIMC-1
  int b = blockIdx.y;
  const float* xb = x + (size_t)b * NSEQ * DIMC + d;
  float s = 0.f, s2 = 0.f;
  for (int n = 0; n < NSEQ; ++n) {
    float v = xb[(size_t)n * DIMC];
    s += v; s2 += v * v;
  }
  float mean = s * (1.0f / NSEQ);
  float var  = s2 * (1.0f / NSEQ) - mean * mean;
  float r    = rsqrtf(fmaxf(var, 1e-5f)) * g[d];
  unsigned short* o = xn + (size_t)b * NSEQ * DIMC + d;
  for (int n = 0; n < NSEQ; ++n)
    o[(size_t)n * DIMC] = f2bf_bits((xb[(size_t)n * DIMC] - mean) * r);
}

// ---------------------------------------------------------------------------
// 2) f32 -> bf16 weight conversion
// ---------------------------------------------------------------------------
__global__ void f32_to_bf16_kernel(const float* __restrict__ in,
                                   unsigned short* __restrict__ out, int n) {
  int i = blockIdx.x * blockDim.x + threadIdx.x;
  if (i < n) out[i] = f2bf_bits(in[i]);
}

// ---------------------------------------------------------------------------
// 3) QKV projection: [4096 x 3072] = xn[4096x1024] * [Wq;Wkv]^T
//    one wave per 32x32 output tile (2x2 WMMA frags), K-loop step 32.
//    Epilogue scatters to q (scaled), k, and transposed vT.
// ---------------------------------------------------------------------------
__global__ __launch_bounds__(32)
void qkv_gemm_kernel(const unsigned short* __restrict__ xn,
                     const unsigned short* __restrict__ wq,
                     const unsigned short* __restrict__ wkv,
                     unsigned short* __restrict__ qw,
                     unsigned short* __restrict__ kw,
                     unsigned short* __restrict__ vT) {
  int lane = threadIdx.x;
  int n0 = blockIdx.x * 32;      // output feature block (0..3071)
  int m0 = blockIdx.y * 32;      // row block (flattened b*NSEQ+n)
  int row = lane & 15;

  f32x8 acc00 = {}, acc01 = {}, acc10 = {}, acc11 = {};

  for (int kk = 0; kk < DIMC; kk += 32) {
    bf16x16 a0 = load_frag(xn + (size_t)(m0 + row) * DIMC + kk, lane);
    bf16x16 a1 = load_frag(xn + (size_t)(m0 + 16 + row) * DIMC + kk, lane);
    int c0 = n0 + row, c1 = n0 + 16 + row;      // B col = weight row
    const unsigned short* w0 = (c0 < INNER) ? wq  + (size_t)c0 * DIMC
                                            : wkv + (size_t)(c0 - INNER) * DIMC;
    const unsigned short* w1 = (c1 < INNER) ? wq  + (size_t)c1 * DIMC
                                            : wkv + (size_t)(c1 - INNER) * DIMC;
    bf16x16 b0 = load_frag(w0 + kk, lane);
    bf16x16 b1 = load_frag(w1 + kk, lane);
    acc00 = wmma_bf16(a0, b0, acc00);
    acc01 = wmma_bf16(a0, b1, acc01);
    acc10 = wmma_bf16(a1, b0, acc10);
    acc11 = wmma_bf16(a1, b1, acc11);
  }

  int halfoff = (lane >> 4) << 3;
  int col = lane & 15;
#pragma unroll
  for (int mt = 0; mt < 2; ++mt) {
#pragma unroll
    for (int nt = 0; nt < 2; ++nt) {
      const f32x8& a = mt ? (nt ? acc11 : acc10) : (nt ? acc01 : acc00);
#pragma unroll
      for (int r = 0; r < 8; ++r) {
        int mg = m0 + mt * 16 + r + halfoff;
        int cg = n0 + nt * 16 + col;
        float v = a[r];
        int b = mg >> 11, n = mg & (NSEQ - 1);
        if (cg < INNER) {                                  // Q (scaled)
          int h = cg >> 6, d = cg & 63;
          qw[(((size_t)(b * HEADS + h) * NSEQ) + n) * DHEAD + d] =
              f2bf_bits(v * 0.125f);                       // DH^-0.5
        } else if (cg < 2 * INNER) {                       // K
          int c = cg - INNER; int h = c >> 6, d = c & 63;
          kw[(((size_t)(b * HEADS + h) * NSEQ) + n) * DHEAD + d] = f2bf_bits(v);
        } else {                                           // V, transposed
          int c = cg - 2 * INNER; int h = c >> 6, d = c & 63;
          vT[(((size_t)(b * HEADS + h) * DHEAD) + d) * NSEQ + n] = f2bf_bits(v);
        }
      }
    }
  }
}

// ---------------------------------------------------------------------------
// 4) Flash attention: one wave per (b,h, 16-row q tile). j-step = 32.
//    S = q*k^T (WMMA) + bias (NT stream), online softmax with 16-lane
//    butterfly reductions, P transposed D-layout -> A-layout via LDS,
//    O += P*V (WMMA, V^T gives contiguous K loads).
// ---------------------------------------------------------------------------
__global__ __launch_bounds__(32)
void attn_kernel(const unsigned short* __restrict__ qw,
                 const unsigned short* __restrict__ kw,
                 const unsigned short* __restrict__ vT,
                 const float* __restrict__ bias,
                 unsigned short* __restrict__ ao) {
  __shared__ unsigned short pls[16 * 32] __attribute__((aligned(16)));

  int lane = threadIdx.x;
  int q0 = blockIdx.x * 16;
  int bh = blockIdx.y;                 // b = bh>>4, h = bh&15
  int h = bh & (HEADS - 1);
  int b = bh >> 4;
  int row = lane & 15;
  int halfoff = (lane >> 4) << 3;
  int col = lane & 15;

  size_t headoff = (size_t)bh * NSEQ * DHEAD;

  bf16x16 qa0 = load_frag(qw + headoff + (size_t)(q0 + row) * DHEAD + 0, lane);
  bf16x16 qa1 = load_frag(qw + headoff + (size_t)(q0 + row) * DHEAD + 32, lane);

  float m[8], l[8];
  f32x8 o0 = {}, o1 = {}, o2 = {}, o3 = {};
#pragma unroll
  for (int r = 0; r < 8; ++r) { m[r] = -3.0e38f; l[r] = 0.f; }

  const float* brow = bias + ((size_t)h * NSEQ + q0) * NSEQ;
  const unsigned short* vhead = vT + (size_t)bh * DHEAD * NSEQ;

  for (int j0 = 0; j0 < NSEQ; j0 += 32) {
    bf16x16 kb00 = load_frag(kw + headoff + (size_t)(j0 + row) * DHEAD + 0, lane);
    bf16x16 kb01 = load_frag(kw + headoff + (size_t)(j0 + row) * DHEAD + 32, lane);
    bf16x16 kb10 = load_frag(kw + headoff + (size_t)(j0 + 16 + row) * DHEAD + 0, lane);
    bf16x16 kb11 = load_frag(kw + headoff + (size_t)(j0 + 16 + row) * DHEAD + 32, lane);

    f32x8 s0 = {}, s1 = {};
    s0 = wmma_bf16(qa0, kb00, s0);
    s0 = wmma_bf16(qa1, kb01, s0);
    s1 = wmma_bf16(qa0, kb10, s1);
    s1 = wmma_bf16(qa1, kb11, s1);

    // bias: streamed non-temporally (256MB tensor dominates BW)
#pragma unroll
    for (int r = 0; r < 8; ++r) {
      const float* bp = brow + (size_t)(r + halfoff) * NSEQ + j0 + col;
      s0[r] += __builtin_nontemporal_load(bp);
      s1[r] += __builtin_nontemporal_load(bp + 16);
    }

    // online softmax: row lives across 16 lanes of a half-wave per VGPR
#pragma unroll
    for (int r = 0; r < 8; ++r) {
      float mx = fmaxf(s0[r], s1[r]);
#pragma unroll
      for (int off = 8; off >= 1; off >>= 1)
        mx = fmaxf(mx, __shfl_xor(mx, off, 32));
      float mn = fmaxf(m[r], mx);
      float sc = __expf(m[r] - mn);
      m[r] = mn;
      float p0 = __expf(s0[r] - mn);
      float p1 = __expf(s1[r] - mn);
      float rs = p0 + p1;
#pragma unroll
      for (int off = 8; off >= 1; off >>= 1)
        rs += __shfl_xor(rs, off, 32);
      l[r] = l[r] * sc + rs;
      o0[r] *= sc; o1[r] *= sc; o2[r] *= sc; o3[r] *= sc;
      pls[(r + halfoff) * 32 + col]      = f2bf_bits(p0);
      pls[(r + halfoff) * 32 + col + 16] = f2bf_bits(p1);
    }
    asm volatile("s_wait_dscnt 0" ::: "memory");   // LDS RAW fence (same wave)

    bf16x16 pa = load_frag(pls + row * 32, lane);  // P in A-fragment layout
    bf16x16 vb0 = load_frag(vhead + (size_t)(0 * 16 + row) * NSEQ + j0, lane);
    bf16x16 vb1 = load_frag(vhead + (size_t)(1 * 16 + row) * NSEQ + j0, lane);
    bf16x16 vb2 = load_frag(vhead + (size_t)(2 * 16 + row) * NSEQ + j0, lane);
    bf16x16 vb3 = load_frag(vhead + (size_t)(3 * 16 + row) * NSEQ + j0, lane);
    o0 = wmma_bf16(pa, vb0, o0);
    o1 = wmma_bf16(pa, vb1, o1);
    o2 = wmma_bf16(pa, vb2, o2);
    o3 = wmma_bf16(pa, vb3, o3);
    asm volatile("" ::: "memory");                 // keep LDS reuse ordered
  }

  // epilogue: normalize rows, write bf16 [b, n, h*64 + d]
#pragma unroll
  for (int r = 0; r < 8; ++r) {
    float inv = 1.0f / l[r];
    size_t rowbase = ((size_t)(b * NSEQ + q0 + r + halfoff)) * INNER + h * DHEAD;
    ao[rowbase + 0 * 16 + col] = f2bf_bits(o0[r] * inv);
    ao[rowbase + 1 * 16 + col] = f2bf_bits(o1[r] * inv);
    ao[rowbase + 2 * 16 + col] = f2bf_bits(o2[r] * inv);
    ao[rowbase + 3 * 16 + col] = f2bf_bits(o3[r] * inv);
  }
}

// ---------------------------------------------------------------------------
// 5) output projection: out[4096x1024] = ao[4096x1024] * Wo^T  (fp32 out)
// ---------------------------------------------------------------------------
__global__ __launch_bounds__(32)
void out_gemm_kernel(const unsigned short* __restrict__ ao,
                     const unsigned short* __restrict__ wo,
                     float* __restrict__ out) {
  int lane = threadIdx.x;
  int n0 = blockIdx.x * 32;
  int m0 = blockIdx.y * 32;
  int row = lane & 15;

  f32x8 acc00 = {}, acc01 = {}, acc10 = {}, acc11 = {};
  for (int kk = 0; kk < INNER; kk += 32) {
    bf16x16 a0 = load_frag(ao + (size_t)(m0 + row) * INNER + kk, lane);
    bf16x16 a1 = load_frag(ao + (size_t)(m0 + 16 + row) * INNER + kk, lane);
    bf16x16 b0 = load_frag(wo + (size_t)(n0 + row) * INNER + kk, lane);
    bf16x16 b1 = load_frag(wo + (size_t)(n0 + 16 + row) * INNER + kk, lane);
    acc00 = wmma_bf16(a0, b0, acc00);
    acc01 = wmma_bf16(a0, b1, acc01);
    acc10 = wmma_bf16(a1, b0, acc10);
    acc11 = wmma_bf16(a1, b1, acc11);
  }

  int halfoff = (lane >> 4) << 3;
  int col = lane & 15;
#pragma unroll
  for (int r = 0; r < 8; ++r) {
    out[(size_t)(m0 + r + halfoff) * DIMC + n0 + col]           = acc00[r];
    out[(size_t)(m0 + r + halfoff) * DIMC + n0 + 16 + col]      = acc01[r];
    out[(size_t)(m0 + 16 + r + halfoff) * DIMC + n0 + col]      = acc10[r];
    out[(size_t)(m0 + 16 + r + halfoff) * DIMC + n0 + 16 + col] = acc11[r];
  }
}

// ---------------------------------------------------------------------------
extern "C" void kernel_launch(void* const* d_in, const int* in_sizes, int n_in,
                              void* d_out, int out_size, void* d_ws, size_t ws_size,
                              hipStream_t stream) {
  const float* x    = (const float*)d_in[0];
  const float* bias = (const float*)d_in[1];
  const float* g    = (const float*)d_in[2];
  const float* Wq   = (const float*)d_in[3];
  const float* Wkv  = (const float*)d_in[4];
  const float* Wo   = (const float*)d_in[5];
  float* out = (float*)d_out;

  unsigned short* ws = (unsigned short*)d_ws;
  const size_t E = (size_t)BATCH * NSEQ * DIMC;      // 4M bf16 elements / region
  unsigned short* xn   = ws;
  unsigned short* qw   = ws + 1 * E;
  unsigned short* kw   = ws + 2 * E;
  unsigned short* vTw  = ws + 3 * E;
  unsigned short* ao   = ws + 4 * E;
  unsigned short* wqb  = ws + 5 * E;
  unsigned short* wkvb = ws + 5 * E + (size_t)INNER * DIMC;
  unsigned short* wob  = ws + 5 * E + (size_t)3 * INNER * DIMC;
  // total ws use: 24M bf16 elements = 48 MiB

  f32_to_bf16_kernel<<<(INNER * DIMC + 255) / 256, 256, 0, stream>>>(Wq, wqb, INNER * DIMC);
  f32_to_bf16_kernel<<<(2 * INNER * DIMC + 255) / 256, 256, 0, stream>>>(Wkv, wkvb, 2 * INNER * DIMC);
  f32_to_bf16_kernel<<<(INNER * DIMC + 255) / 256, 256, 0, stream>>>(Wo, wob, INNER * DIMC);

  stats_norm_kernel<<<dim3(DIMC / 256, BATCH), 256, 0, stream>>>(x, g, xn);

  qkv_gemm_kernel<<<dim3(3 * INNER / 32, BATCH * NSEQ / 32), 32, 0, stream>>>(
      xn, wqb, wkvb, qw, kw, vTw);

  attn_kernel<<<dim3(NSEQ / 16, BATCH * HEADS), 32, 0, stream>>>(
      qw, kw, vTw, bias, ao);

  out_gemm_kernel<<<dim3(DIMC / 32, BATCH * NSEQ / 32), 32, 0, stream>>>(ao, wob, out);
}
